// EnhancedPolarAttention_6983616823899
// MI455X (gfx1250) — matrix operand order repaired and checked by reading, not compiled, not run on hardware
//
#include <hip/hip_runtime.h>
#include <hip/hip_bf16.h>

typedef _Float16 f16;
typedef __attribute__((ext_vector_type(16))) _Float16 v16h;
typedef __attribute__((ext_vector_type(8)))  _Float16 v8h;
typedef __attribute__((ext_vector_type(2)))  _Float16 v2h;
typedef __attribute__((ext_vector_type(8)))  float    v8f;
typedef __attribute__((ext_vector_type(4)))  float    v4f;

#define N_POS 4096
#define C_IN  128
#define KD    256
#define NH    8
#define HD    32
#define AD    64   // augmented head dim: [cos-block | sin-block]

static __device__ __forceinline__ v8f wmma_f16(v16h a, v16h b, v8f c) {
  // D = A(16x32 f16) * B(32x16 f16) + C(f32)
  return __builtin_amdgcn_wmma_f32_16x16x32_f16(false, a, false, b, (short)0, c, false, false);
}

// A-operand fragment: 16 rows x 32 K. src row-major, ld in halves.
// lane: row = row0 + lane%16 ; kb = 16*(lane/16); v0-3: K=k0+kb..+7 ; v4-7: K=k0+kb+8..+15
static __device__ __forceinline__ v16h load_fragA(const f16* __restrict__ base, int row0,
                                                  int ld, int k0, int lane) {
  int r  = row0 + (lane & 15);
  int kb = (lane >> 4) << 4;
  const f16* p = base + r * ld + k0 + kb;
  v8h lo = *(const v8h*)(p);
  v8h hi = *(const v8h*)(p + 8);
  v16h out;
#pragma unroll
  for (int i = 0; i < 8; ++i) { out[i] = lo[i]; out[8 + i] = hi[i]; }
  return out;
}

// B-operand fragment: 32 K x 16 cols, where src row n = output column (contiguous K).
// lane: col = col0 + lane%16 ; K range = k0 + 16*(lane/16) .. +15 (VGPR j holds K pair 2j,2j+1)
static __device__ __forceinline__ v16h load_fragB(const f16* __restrict__ base, int col0,
                                                  int ld, int k0, int lane) {
  int n  = col0 + (lane & 15);
  int kk = k0 + ((lane >> 4) << 4);
  const f16* p = base + n * ld + kk;
  v8h lo = *(const v8h*)(p);
  v8h hi = *(const v8h*)(p + 8);
  v16h out;
#pragma unroll
  for (int i = 0; i < 8; ++i) { out[i] = lo[i]; out[8 + i] = hi[i]; }
  return out;
}

// ---------------- prep kernels ----------------
__global__ void k_cvt_f16(const float* __restrict__ src, f16* __restrict__ dst, int n) {
  int i = blockIdx.x * blockDim.x + threadIdx.x;
  if (i < n) dst[i] = (f16)src[i];
}

__global__ void k_transpose_f16(const float* __restrict__ src, f16* __restrict__ dst,
                                int rows, int cols) {
  int i = blockIdx.x * blockDim.x + threadIdx.x;
  if (i < rows * cols) {
    int r = i / cols, c = i % cols;
    dst[c * rows + r] = (f16)src[i];
  }
}

__global__ void k_polar(float* __restrict__ wck, float* __restrict__ wsk,
                        float* __restrict__ cqa, float* __restrict__ sqa) {
  int p = blockIdx.x * blockDim.x + threadIdx.x;
  if (p >= N_POS) return;
  float xx = (float)(p >> 6);   // matches reference meshgrid flatten: x = p // H
  float yy = (float)(p & 63);   // y = p % H
  float dx = xx - 32.0f, dy = yy - 32.0f;
  float r = sqrtf(dx * dx + dy * dy) + 1e-6f;
  float theta = atan2f(dy, dx);
  float rmax = sqrtf(2048.0f) + 1e-6f;       // corner (0,0): 32*sqrt(2)
  float log_r = logf(r) / logf(rmax);
  float w = 1.0f / (log_r + 1.0f);
  float scale = 0.17677669529663687f;        // 1/sqrt(32)
  float ws = w * scale;
  float ct = cosf(theta), st = sinf(theta);
  wck[p] = ws * ct;
  wsk[p] = ws * st;
  cqa[p] = ct;
  sqa[p] = st;
}

// ---------------- QKV projection ----------------
// Q,K columns computed transposed: D = Wp^T (o x cin) * X^T (cin x n) -> tile rows = o, cols = n.
// Augmented packed stores: Qa/Ka [head][n][64] = [mod0*val | mod1*val].
__global__ __launch_bounds__(256) void k_proj_qk(const f16* __restrict__ Wpt,
                                                 const f16* __restrict__ Xh,
                                                 const float* __restrict__ bp,
                                                 const float* __restrict__ wck,
                                                 const float* __restrict__ wsk,
                                                 const float* __restrict__ cqa,
                                                 const float* __restrict__ sqa,
                                                 f16* __restrict__ Qa, f16* __restrict__ Ka) {
  int wave = (blockIdx.x * blockDim.x + threadIdx.x) >> 5;  // 8192 waves
  int lane = threadIdx.x & 31;
  int o0 = (wave >> 8) << 4;   // 0..511 step 16 (q then k columns)
  int n0 = (wave & 255) << 4;  // 0..4095 step 16
  v8f acc = {};
#pragma unroll
  for (int t = 0; t < 4; ++t) {
    v16h a = load_fragA(Wpt, o0, C_IN, t * 32, lane);
    v16h b = load_fragB(Xh, n0, C_IN, t * 32, lane);
    acc = wmma_f16(a, b, acc);
  }
  int hh = lane >> 4;
  int rb = o0 + (hh << 3);          // bias row base: o = rb + i
  int n  = n0 + (lane & 15);
  f16* dst; int head, d0;
  float m0v, m1v;                   // per-position modulation scalars
  if (o0 < 256) {
    dst = Qa; head = o0 >> 5; d0 = o0 & 31;
    m0v = cqa[n]; m1v = sqa[n];
  } else {
    dst = Ka; head = (o0 - 256) >> 5; d0 = (o0 - 256) & 31;
    m0v = wck[n]; m1v = wsk[n];     // includes 1/sqrt(hd) and radial weight
  }
  v8h ov0, ov1;
#pragma unroll
  for (int i = 0; i < 8; ++i) {
    float v = acc[i] + bp[rb + i];
    ov0[i] = (f16)(v * m0v);
    ov1[i] = (f16)(v * m1v);
  }
  f16* rowp = dst + (head * N_POS + n) * AD + d0 + (hh << 3);
  *(v8h*)(rowp)      = ov0;   // cos block: d
  *(v8h*)(rowp + 32) = ov1;   // sin block: 32 + d
}

// V columns computed straight: D = X (n x cin) * Wp (cin x o); packed stores into Vt [head][d][n].
__global__ __launch_bounds__(256) void k_proj_v(const f16* __restrict__ Xh,
                                                const f16* __restrict__ Wpt,
                                                const float* __restrict__ bp,
                                                f16* __restrict__ Vt) {
  int wave = (blockIdx.x * blockDim.x + threadIdx.x) >> 5;  // 4096 waves
  int lane = threadIdx.x & 31;
  int n0  = (wave >> 4) << 4;
  int oc0 = 512 + ((wave & 15) << 4);
  v8f acc = {};
#pragma unroll
  for (int t = 0; t < 4; ++t) {
    v16h a = load_fragA(Xh, n0, C_IN, t * 32, lane);
    v16h b = load_fragB(Wpt, oc0, C_IN, t * 32, lane);
    acc = wmma_f16(a, b, acc);
  }
  int oc = oc0 + (lane & 15);
  float bias = bp[oc];
  int head = (oc - 512) >> 5;
  int d    = (oc - 512) & 31;
  v8h outv;
#pragma unroll
  for (int i = 0; i < 8; ++i) outv[i] = (f16)(acc[i] + bias);
  *(v8h*)(Vt + (head * HD + d) * N_POS + n0 + ((lane >> 4) << 3)) = outv;
}

// ---------------- flash attention (register resident, S^T formulation) ----------------
// Scores come fully modulated out of the WMMA thanks to the 64-dim augmentation.
__global__ __launch_bounds__(256) void k_attn(const f16* __restrict__ Qa, const f16* __restrict__ Ka,
                                              const f16* __restrict__ Vt,
                                              f16* __restrict__ O) {
  int wave = (blockIdx.x * blockDim.x + threadIdx.x) >> 5;  // 2048 waves: 8 heads x 256 q-tiles
  int lane = threadIdx.x & 31;
  int head = wave >> 8;
  int m0   = (wave & 255) << 4;
  const f16* Q = Qa + head * N_POS * AD;
  const f16* K = Ka + head * N_POS * AD;
  const f16* V = Vt + head * HD * N_POS;
  int qi  = m0 + (lane & 15);
  int h   = lane >> 4;
  int kb8 = h << 3;
  v16h bq0 = load_fragB(Q, m0, AD, 0, lane);   // B operand: Q~^T, k-dims 0..31
  v16h bq1 = load_fragB(Q, m0, AD, 32, lane);  // k-dims 32..63
  v8f acc0 = {}, acc1 = {};                    // O^T fragments: d 0-15 / 16-31
  float mrow = -3.0e38f, lsum = 0.0f;
#pragma unroll 1
  for (int n = 0; n < N_POS; n += 32) {
    if (n + 32 < N_POS) {
      __builtin_prefetch(K + (n + 32 + lane) * AD, 0, 3);     // next 32 key rows
      __builtin_prefetch(V + lane * N_POS + n + 32, 0, 3);    // next V^T columns
    }
    v16h ka00 = load_fragA(K, n, AD, 0, lane);
    v16h ka01 = load_fragA(K, n, AD, 32, lane);
    v16h ka10 = load_fragA(K, n + 16, AD, 0, lane);
    v16h ka11 = load_fragA(K, n + 16, AD, 32, lane);
    v8f z = {};
    v8f s0 = wmma_f16(ka00, bq0, z);   // S^T rows = keys n..n+15 (modulated)
    s0 = wmma_f16(ka01, bq1, s0);
    v8f s1 = wmma_f16(ka10, bq0, z);   // keys n+16..n+31
    s1 = wmma_f16(ka11, bq1, s1);
    float tm = -3.0e38f;
#pragma unroll
    for (int i = 0; i < 8; ++i) tm = fmaxf(tm, fmaxf(s0[i], s1[i]));
    tm = fmaxf(tm, __shfl_xor(tm, 16, 32));     // combine key-halves per query column
    float mnew  = fmaxf(mrow, tm);
    float alpha = __expf(mrow - mnew);
    mrow = mnew;
    float p0[8], p1[8];
    float tsum = 0.0f;
#pragma unroll
    for (int i = 0; i < 8; ++i) {
      p0[i] = __expf(s0[i] - mnew);
      p1[i] = __expf(s1[i] - mnew);
      tsum += p0[i] + p1[i];
    }
    tsum += __shfl_xor(tsum, 16, 32);
    lsum = lsum * alpha + tsum;
#pragma unroll
    for (int i = 0; i < 8; ++i) { acc0[i] *= alpha; acc1[i] *= alpha; }
    // Build P^T B-fragment: lane-local f16 packs + one half-wave swap (xor 16).
    unsigned q0[4], q1[4], sendv[4], recvv[4];
#pragma unroll
    for (int j = 0; j < 4; ++j) {
      v2h t0; t0[0] = (f16)p0[2 * j]; t0[1] = (f16)p0[2 * j + 1];
      v2h t1; t1[0] = (f16)p1[2 * j]; t1[1] = (f16)p1[2 * j + 1];
      q0[j] = __builtin_bit_cast(unsigned, t0);
      q1[j] = __builtin_bit_cast(unsigned, t1);
      sendv[j] = h ? q0[j] : q1[j];
    }
#pragma unroll
    for (int j = 0; j < 4; ++j)
      recvv[j] = (unsigned)__shfl_xor((int)sendv[j], 16, 32);
    union { v16h v; unsigned u[8]; } pb;
#pragma unroll
    for (int j = 0; j < 4; ++j) {
      pb.u[j]     = h ? recvv[j] : q0[j];    // K = 16h+0..7  of P[q, .]
      pb.u[4 + j] = h ? q1[j]    : recvv[j]; // K = 16h+8..15 of P[q, .]
    }
    v16h va0 = load_fragA(V, 0, N_POS, n, lane);   // V^T rows d=0..15, keys n..n+31
    v16h va1 = load_fragA(V, 16, N_POS, n, lane);  // d=16..31
    acc0 = wmma_f16(va0, pb.v, acc0);
    acc1 = wmma_f16(va1, pb.v, acc1);
  }
  float inv = 1.0f / lsum;
  v8h o0v, o1v;
#pragma unroll
  for (int i = 0; i < 8; ++i) {
    o0v[i] = (f16)(acc0[i] * inv);
    o1v[i] = (f16)(acc1[i] * inv);
  }
  // O^T fragment (d = i + 8h, q fixed per lane) -> packed row-major store O[q][head*32+d]
  f16* orow = O + qi * KD + head * HD;
  *(v8h*)(orow + kb8)      = o0v;
  *(v8h*)(orow + 16 + kb8) = o1v;
}

// ---------------- final projection: out = O @ Wf + bf (f32 output) ----------------
__global__ __launch_bounds__(256) void k_out(const f16* __restrict__ O, const f16* __restrict__ Wft,
                                             const float* __restrict__ bf, float* __restrict__ out) {
  int wave = (blockIdx.x * blockDim.x + threadIdx.x) >> 5;  // 4096 waves
  int lane = threadIdx.x & 31;
  int n0 = (wave >> 4) << 4;
  int o0 = (wave & 15) << 4;
  v8f acc = {};
#pragma unroll
  for (int t = 0; t < 8; ++t) {
    v16h a = load_fragA(O, n0, KD, t * 32, lane);
    v16h b = load_fragB(Wft, o0, KD, t * 32, lane);
    acc = wmma_f16(a, b, acc);
  }
  int oc = o0 + (lane & 15);
  float bias = bf[oc];
  int rbase = n0 + ((lane >> 4) << 3);
#pragma unroll
  for (int i = 0; i < 8; ++i)
    out[(rbase + i) * KD + oc] = acc[i] + bias;
}

extern "C" void kernel_launch(void* const* d_in, const int* in_sizes, int n_in,
                              void* d_out, int out_size, void* d_ws, size_t ws_size,
                              hipStream_t stream) {
  (void)in_sizes; (void)n_in; (void)out_size; (void)ws_size;
  const float* x  = (const float*)d_in[0];
  const float* Wp = (const float*)d_in[1];
  const float* bp = (const float*)d_in[2];
  const float* Wf = (const float*)d_in[3];
  const float* bf = (const float*)d_in[4];
  float* out = (float*)d_out;

  char* ws = (char*)d_ws;
  f16*   Xh  = (f16*)(ws + 0);           // 4096x128 f16             (1,048,576 B)
  f16*   Wpt = (f16*)(ws + 1048576);     // 768x128 f16              (  196,608 B)
  f16*   Wft = (f16*)(ws + 1245184);     // 256x256 f16              (  131,072 B)
  f16*   Qa  = (f16*)(ws + 1376256);     // 8 x 4096 x 64 f16        (4,194,304 B)
  f16*   Ka  = (f16*)(ws + 5570560);     // 8 x 4096 x 64 f16        (4,194,304 B)
  f16*   Vt  = (f16*)(ws + 9764864);     // 8 x 32 x 4096 f16        (2,097,152 B)
  f16*   Ob  = (f16*)(ws + 11862016);    // 4096 x 256 f16           (2,097,152 B)
  float* wck = (float*)(ws + 13959168);  // 4096 f32
  float* wsk = (float*)(ws + 13975552);
  float* cqa = (float*)(ws + 13991936);
  float* sqa = (float*)(ws + 14008320);  // end: 14,024,704 B

  k_polar<<<16, 256, 0, stream>>>(wck, wsk, cqa, sqa);
  k_cvt_f16<<<2048, 256, 0, stream>>>(x, Xh, N_POS * C_IN);
  k_transpose_f16<<<384, 256, 0, stream>>>(Wp, Wpt, C_IN, 3 * KD);
  k_transpose_f16<<<256, 256, 0, stream>>>(Wf, Wft, KD, KD);
  k_proj_qk<<<1024, 256, 0, stream>>>(Wpt, Xh, bp, wck, wsk, cqa, sqa, Qa, Ka); // 8192 waves
  k_proj_v<<<512, 256, 0, stream>>>(Xh, Wpt, bp, Vt);        // 4096 waves
  k_attn<<<256, 256, 0, stream>>>(Qa, Ka, Vt, Ob);           // 2048 waves
  k_out<<<512, 256, 0, stream>>>(Ob, Wft, bf, out);          // 4096 waves
}
